// MoELayer_30537217474766
// MI455X (gfx1250) — matrix operand order repaired and checked by reading, not compiled, not run on hardware
//
#include <hip/hip_runtime.h>
#include <hip/hip_bf16.h>
#include <math.h>
#include <stdint.h>

typedef _Float16 half_t;
typedef __attribute__((ext_vector_type(16))) _Float16 v16h;
typedef __attribute__((ext_vector_type(8)))  _Float16 v8h;
typedef __attribute__((ext_vector_type(8)))  float    v8f;

#define D_MODEL     1024
#define NUM_EXPERTS 8
#define D_FF        1024
#define NTOK        8192
#define CAP         1024
#define THRESH      0.9f

// ---------------- 1) fp32 -> f16 conversion of x, w1, w2 ----------------
__global__ void moe_cvt(const float* __restrict__ x,
                        const float* __restrict__ w1,
                        const float* __restrict__ w2,
                        half_t* __restrict__ xb,
                        half_t* __restrict__ w1b,
                        half_t* __restrict__ w2b) {
    const size_t NX = (size_t)NTOK * D_MODEL;
    const size_t NW = (size_t)NUM_EXPERTS * D_FF * D_MODEL;
    const size_t total = NX + 2 * NW;
    size_t stride = (size_t)gridDim.x * blockDim.x;
    for (size_t i = (size_t)blockIdx.x * blockDim.x + threadIdx.x; i < total; i += stride) {
        if (i < NX)            xb[i]            = (half_t)x[i];
        else if (i < NX + NW)  w1b[i - NX]      = (half_t)w1[i - NX];
        else                   w2b[i - NX - NW] = (half_t)w2[i - NX - NW];
    }
}

// ---------------- 2) gating: softmax + threshold top-p + rank score R ----------------
__global__ void moe_gate_route(const float* __restrict__ x,
                               const float* __restrict__ gw,
                               float* __restrict__ probs,
                               float* __restrict__ Rarr) {
    int wave = threadIdx.x >> 5;
    int lane = threadIdx.x & 31;
    int n = blockIdx.x * 8 + wave;
    float acc[NUM_EXPERTS];
#pragma unroll
    for (int e = 0; e < NUM_EXPERTS; ++e) acc[e] = 0.f;
    const float* xr = x + (size_t)n * D_MODEL;
    for (int d = lane; d < D_MODEL; d += 32) {
        float xv = xr[d];
#pragma unroll
        for (int e = 0; e < NUM_EXPERTS; ++e) acc[e] += xv * gw[e * D_MODEL + d];
    }
#pragma unroll
    for (int e = 0; e < NUM_EXPERTS; ++e)
        for (int off = 16; off > 0; off >>= 1)
            acc[e] += __shfl_xor(acc[e], off, 32);
    if (lane == 0) {
        float m = acc[0];
        for (int e = 1; e < NUM_EXPERTS; ++e) m = fmaxf(m, acc[e]);
        float p[NUM_EXPERTS], s = 0.f;
        for (int e = 0; e < NUM_EXPERTS; ++e) { p[e] = __expf(acc[e] - m); s += p[e]; }
        float inv = 1.f / s;
        for (int e = 0; e < NUM_EXPERTS; ++e) { p[e] *= inv; probs[n * 8 + e] = p[e]; }
        int   idx[NUM_EXPERTS]; float ps[NUM_EXPERTS];
        for (int e = 0; e < NUM_EXPERTS; ++e) { idx[e] = e; ps[e] = p[e]; }
        for (int i = 1; i < NUM_EXPERTS; ++i) {   // stable descending insertion sort
            float pv = ps[i]; int iv = idx[i]; int j = i - 1;
            while (j >= 0 && ps[j] < pv) { ps[j+1] = ps[j]; idx[j+1] = idx[j]; --j; }
            ps[j+1] = pv; idx[j+1] = iv;
        }
        int k = NUM_EXPERTS - 1; float cs = 0.f;
        for (int j = 0; j < NUM_EXPERTS; ++j) { cs += ps[j]; if (cs >= THRESH) { k = j; break; } }
        float R[NUM_EXPERTS];
        for (int e = 0; e < NUM_EXPERTS; ++e) R[e] = -1e9f;
        for (int j = 0; j <= k; ++j) R[idx[j]] = ps[j] - (float)(j + 1);
        for (int e = 0; e < NUM_EXPERTS; ++e) Rarr[n * 8 + e] = R[e];
    }
}

// ---------------- 3) per-expert top-C keep via rank counting ----------------
__global__ void moe_rank_keep(const float* __restrict__ Rarr,
                              const float* __restrict__ probs,
                              float* __restrict__ scoreSel) {
    __shared__ float rs[NTOK];
    int e = blockIdx.y;
    for (int i = threadIdx.x; i < NTOK; i += blockDim.x)
        rs[i] = Rarr[i * 8 + e];
    __syncthreads();
    int n = blockIdx.x * blockDim.x + threadIdx.x;
    float rn = rs[n];
    int cntgt = 0;
#pragma unroll 8
    for (int m = 0; m < NTOK; ++m) {
        float rm = rs[m];
        cntgt += (int)((rm > rn) || ((rm == rn) && (m < n)));
    }
    bool keep = cntgt < CAP;
    scoreSel[n * 8 + e] = (keep && rn > -5e8f) ? probs[n * 8 + e] : 0.f;
}

// ---------------- 4) deterministic per-expert compaction (LDS prefix scan) ----------------
__global__ void moe_build_lists(const float* __restrict__ scoreSel,
                                int* __restrict__ list,
                                int* __restrict__ slot,
                                int* __restrict__ cnt) {
    __shared__ int sc[1024];
    int e = blockIdx.x;
    int t = threadIdx.x;
    int base = t * 8;
    int mask = 0, c = 0;
#pragma unroll
    for (int j = 0; j < 8; ++j) {
        bool f = scoreSel[(size_t)(base + j) * 8 + e] > 0.f;
        mask |= (f ? 1 : 0) << j;
        c += (int)f;
    }
    sc[t] = c;
    __syncthreads();
    for (int off = 1; off < 1024; off <<= 1) {
        int v = (t >= off) ? sc[t - off] : 0;
        __syncthreads();
        sc[t] += v;
        __syncthreads();
    }
    int pos = sc[t] - c;  // exclusive prefix
#pragma unroll
    for (int j = 0; j < 8; ++j) {
        int n = base + j;
        if (mask & (1 << j)) {
            list[e * CAP + pos] = n;
            slot[n * 8 + e] = pos;
            ++pos;
        } else {
            slot[n * 8 + e] = -1;
        }
    }
    if (t == 1023) cnt[e] = sc[1023];
}

// ---------------- 5/6) tiled expert GEMM via WMMA f16 + async-to-LDS A staging ----------------
// Block = 256 threads = 8 waves (2M x 4N). Block tile 64M x 256N; wave tile 32M x 64N
// (2x4 fragments, 8 v8f accumulators). K-step 32; A slice (64x32 f16) double-buffered in
// LDS via global_load_async_to_lds_b128 + s_wait_asynccnt. All 4 B fragments are loaded
// into distinct registers before the WMMA burst so the backend keeps 4 loads in flight.
template<bool FFN1>
__global__ void moe_gemm_tiled(const half_t* __restrict__ Asrc,   // xb (FFN1) or Hbuf
                               const half_t* __restrict__ W,      // w1b or w2b  [1024 x 1024]
                               const int* __restrict__ list,
                               const int* __restrict__ cnt,
                               const float* __restrict__ scoreSel,
                               half_t* __restrict__ Hout,
                               float* __restrict__ Yout) {
    constexpr int K = 1024;
    __shared__ __align__(32) half_t As[2][64][40];   // 80B row pitch: conflict-free b128 reads
    constexpr unsigned BUFSTRIDE = 64u * 40u * (unsigned)sizeof(half_t);

    int e = blockIdx.z;
    int cn = cnt[e];
    int m_base = blockIdx.y * 64;
    if (m_base >= cn) return;                         // block-uniform exit

    int tid = threadIdx.x;
    int wave = tid >> 5, lane = tid & 31;
    int wm = wave >> 2, wn = wave & 3;                // 2 x 4 wave grid
    int r16 = lane & 15;
    bool hi = lane >= 16;

    // ---- async A staging assignment: thread -> (row, 16B segment) of the 64x32 slice
    int srow = tid >> 2;                              // 0..63
    int seg  = tid & 3;                               // 0..3
    int gr = m_base + srow; gr = (gr < cn) ? gr : cn - 1;   // clamped pad rows (discarded at store)
    const half_t* gsrc;
    if constexpr (FFN1) {
        int tokr = list[e * CAP + gr];
        gsrc = Asrc + (size_t)tokr * K + seg * 8;
    } else {
        gsrc = Asrc + ((size_t)e * CAP + gr) * K + seg * 8;
    }
    unsigned lds0 = (unsigned)(uintptr_t)(&As[0][srow][seg * 8]);

    // ---- B fragment pointers: lane = output column, 16 contiguous K halves (lanes>=16: +16K)
    int nblk = blockIdx.x * 256;
    const half_t* bptr[4];
#pragma unroll
    for (int j = 0; j < 4; ++j) {
        int col = nblk + wn * 64 + j * 16 + r16;
        bptr[j] = W + ((size_t)e * 1024 + col) * K + (hi ? 16 : 0);
    }

    v8f c[2][4];
#pragma unroll
    for (int i = 0; i < 2; ++i)
#pragma unroll
        for (int j = 0; j < 4; ++j)
            c[i][j] = v8f{0.f,0.f,0.f,0.f,0.f,0.f,0.f,0.f};

    // prologue: stage K-slice 0 into buffer 0
    asm volatile("global_load_async_to_lds_b128 %0, %1, off"
                 :: "v"(lds0), "v"((unsigned long long)(uintptr_t)gsrc) : "memory");
    asm volatile("s_wait_asynccnt 0x0" ::: "memory");
    __syncthreads();

    int buf = 0;
    for (int k0 = 0; k0 < K; k0 += 32) {
        bool more = (k0 + 32) < K;
        if (more) {   // uniform branch: issue next K-slice into the other buffer
            unsigned l = lds0 + (unsigned)(buf ^ 1) * BUFSTRIDE;
            asm volatile("global_load_async_to_lds_b128 %0, %1, off"
                         :: "v"(l), "v"((unsigned long long)(uintptr_t)(gsrc + k0 + 32)) : "memory");
        }
        // load ALL 4 B fragments first (distinct registers -> 4 loads in flight)
        v16h b[4];
#pragma unroll
        for (int j = 0; j < 4; ++j) {
            __builtin_prefetch(bptr[j] + k0 + 256, 0, 1);   // global_prefetch_b8
            b[j] = *(const v16h*)(bptr[j] + k0);
        }
        // A fragments from LDS: lane(0-15)=row M, K {0..15}; lane(16-31): K {16..31}
        v16h a[2];
#pragma unroll
        for (int i = 0; i < 2; ++i) {
            const half_t* lp = &As[buf][wm * 32 + i * 16 + r16][hi ? 16 : 0];
            v8h a0 = *(const v8h*)lp;
            v8h a1 = *(const v8h*)(lp + 8);
            a[i] = __builtin_shufflevector(a0, a1, 0,1,2,3,4,5,6,7,8,9,10,11,12,13,14,15);
        }
        // 8 WMMAs, independent accumulators (no D->A/B hazards)
#pragma unroll
        for (int j = 0; j < 4; ++j)
#pragma unroll
            for (int i = 0; i < 2; ++i)
                c[i][j] = __builtin_amdgcn_wmma_f32_16x16x32_f16(
                    false, a[i], false, b[j], (short)0, c[i][j], false, false);

        if (more) asm volatile("s_wait_asynccnt 0x0" ::: "memory");
        __syncthreads();
        buf ^= 1;
    }

    // ---- epilogue: C layout c[..][ii] -> (M = base + ii + (hi?8:0), N = colbase + (lane&15))
#pragma unroll
    for (int i = 0; i < 2; ++i) {
        int m0 = m_base + wm * 32 + i * 16 + (hi ? 8 : 0);
#pragma unroll
        for (int j = 0; j < 4; ++j) {
            int col = nblk + wn * 64 + j * 16 + r16;
#pragma unroll
            for (int ii = 0; ii < 8; ++ii) {
                int mrow = m0 + ii;
                if (mrow < cn) {
                    float v = c[i][j][ii];
                    if constexpr (FFN1) {
                        float g = 0.5f * v * (1.0f + erff(v * 0.70710678f));  // exact GELU
                        Hout[((size_t)e * CAP + mrow) * 1024 + col] = (half_t)g;
                    } else {
                        int tok = list[e * CAP + mrow];
                        float s = scoreSel[(size_t)tok * 8 + e];
                        Yout[((size_t)e * CAP + mrow) * 1024 + col] = v * s;
                    }
                }
            }
        }
    }
}

// ---------------- 7) deterministic per-token gather over experts ----------------
__global__ void moe_gather(const float* __restrict__ Yout,
                           const int* __restrict__ slot,
                           float* __restrict__ y) {
    size_t idx = (size_t)blockIdx.x * blockDim.x + threadIdx.x;
    int n = (int)(idx >> 10);
    int d = (int)(idx & 1023);
    float acc = 0.f;
#pragma unroll
    for (int e = 0; e < NUM_EXPERTS; ++e) {
        int s = slot[(size_t)n * 8 + e];
        if (s >= 0) acc += Yout[(((size_t)e * CAP + s) << 10) + d];
    }
    y[idx] = acc;
}

// ---------------- host launcher ----------------
extern "C" void kernel_launch(void* const* d_in, const int* in_sizes, int n_in,
                              void* d_out, int out_size, void* d_ws, size_t ws_size,
                              hipStream_t stream) {
    const float* x  = (const float*)d_in[0];
    const float* gw = (const float*)d_in[1];
    const float* w1 = (const float*)d_in[2];
    const float* w2 = (const float*)d_in[3];
    float* y = (float*)d_out;

    const size_t NX = (size_t)NTOK * D_MODEL;
    const size_t NW = (size_t)NUM_EXPERTS * D_FF * D_MODEL;

    char* ws = (char*)d_ws;
    size_t off = 0;
    auto alloc = [&](size_t bytes) -> void* {
        void* p = ws + off;
        off += (bytes + 255) & ~(size_t)255;
        return p;
    };
    half_t* xb      = (half_t*)alloc(NX * sizeof(half_t));
    half_t* w1b     = (half_t*)alloc(NW * sizeof(half_t));
    half_t* w2b     = (half_t*)alloc(NW * sizeof(half_t));
    float*  probs   = (float*) alloc((size_t)NTOK * 8 * sizeof(float));
    float*  Rarr    = (float*) alloc((size_t)NTOK * 8 * sizeof(float));
    float*  scores  = (float*) alloc((size_t)NTOK * 8 * sizeof(float));
    int*    slot    = (int*)   alloc((size_t)NTOK * 8 * sizeof(int));
    int*    list    = (int*)   alloc((size_t)NUM_EXPERTS * CAP * sizeof(int));
    int*    cnt     = (int*)   alloc(256);
    half_t* Hbuf    = (half_t*)alloc((size_t)NUM_EXPERTS * CAP * D_FF * sizeof(half_t));
    float*  Yout    = (float*) alloc((size_t)NUM_EXPERTS * CAP * D_MODEL * sizeof(float));
    (void)ws_size; (void)in_sizes; (void)n_in; (void)out_size;

    moe_cvt<<<4096, 256, 0, stream>>>(x, w1, w2, xb, w1b, w2b);
    moe_gate_route<<<NTOK / 8, 256, 0, stream>>>(x, gw, probs, Rarr);
    moe_rank_keep<<<dim3(NTOK / 256, NUM_EXPERTS), 256, 0, stream>>>(Rarr, probs, scores);
    moe_build_lists<<<NUM_EXPERTS, 1024, 0, stream>>>(scores, list, slot, cnt);
    moe_gemm_tiled<true ><<<dim3(D_FF   / 256, CAP / 64, NUM_EXPERTS), 256, 0, stream>>>(
        xb,   w1b, list, cnt, nullptr, Hbuf, nullptr);
    moe_gemm_tiled<false><<<dim3(D_MODEL / 256, CAP / 64, NUM_EXPERTS), 256, 0, stream>>>(
        Hbuf, w2b, list, cnt, scores,  nullptr, Yout);
    moe_gather<<<(NTOK * D_MODEL) / 256, 256, 0, stream>>>(Yout, slot, y);
}
